// FuXiLinearSequenceModel_66580583023026
// MI455X (gfx1250) — compile-verified, matrix operand order checked by reading, not run on hardware
//
#include <hip/hip_runtime.h>
#include <math.h>

// ---------------- problem constants (fixed by the reference) ----------------
constexpr int B = 32, T = 200, D = 256, H = 4, DH = 32, INNER = 128;
constexpr int L = 8, FF = 1024, NI = 100000, NB = 64;
constexpr int BT = B * T;
constexpr int PROJ = D + 3 * INNER;   // 640
constexpr int CAT  = 3 * D;           // 768
constexpr int UH   = 1024, UIN = 3 * D;

typedef __attribute__((ext_vector_type(16))) __bf16 v16bf;
typedef __attribute__((ext_vector_type(8)))  __bf16 v8bf;
typedef __attribute__((ext_vector_type(8)))  float  v8f;

__device__ __forceinline__ float geluf(float x) {
  return 0.5f * x * (1.0f + erff(x * 0.70710678118654752f));
}
__device__ __forceinline__ float sigmoidf(float x) {
  return 1.0f / (1.0f + expf(-x));
}
__device__ __forceinline__ int imax(int a, int b) { return a > b ? a : b; }

// load one 16-lane-wide bf16 fragment half-pair: elements [0..8) from p,
// [8..16) from p+16 (per the CDNA5 16-bit A/B VGPR layout)
__device__ __forceinline__ v16bf ldfrag(const __bf16* p) {
  v8bf lo = *(const v8bf*)p;
  v8bf hi = *(const v8bf*)(p + 16);
  v16bf r;
#pragma unroll
  for (int i = 0; i < 8; ++i) { r[i] = lo[i]; r[i + 8] = hi[i]; }
  return r;
}

// ---------------- WMMA bf16 GEMM: out = act(A[M,K] @ Wt^T + bias) -----------
// A: bf16 row-major (lda = K), Wt: bf16 N-major (Wt[n*K + k]).
// Optional fp32 output C and/or bf16 output Cbf, both at [m*ldc + n0 + n].
// One 16x64 tile per wave32: one A fragment reused across 4 B fragments.
__global__ void __launch_bounds__(256) k_gemm(
    const __bf16* __restrict__ A, const __bf16* __restrict__ Wt,
    const float* __restrict__ bias, float* __restrict__ C,
    __bf16* __restrict__ Cbf, int M, int N, int K, int ldc, int n0, int act)
{
  const int lane = threadIdx.x & 31;
  const int wid  = blockIdx.x * 8 + (threadIdx.x >> 5);
  const int tilesN  = N >> 4;
  const int tilesN4 = (tilesN + 3) >> 2;
  const int tilesM  = M >> 4;
  const int tM  = wid / tilesN4;
  const int tN4 = wid - tM * tilesN4;
  if (tM >= tilesM) return;

  const int l15  = lane & 15;
  const int koff = (lane >> 4) << 3;  // lanes 16-31 handle K base +8 per ISA layout
  const __bf16* a0 = A + (size_t)(tM * 16 + l15) * K + koff;
  const __bf16* bbase = Wt + (size_t)l15 * K + koff;

  bool valid[4];
  const __bf16* b0[4];
#pragma unroll
  for (int j = 0; j < 4; ++j) {
    int nt = tN4 * 4 + j;
    valid[j] = (nt < tilesN);
    b0[j] = bbase + (size_t)(valid[j] ? nt : 0) * 16 * K;
  }

  v8f acc[4];
#pragma unroll
  for (int j = 0; j < 4; ++j) {
    v8f z = {0.f, 0.f, 0.f, 0.f, 0.f, 0.f, 0.f, 0.f};
    acc[j] = z;
  }

  for (int k = 0; k < K; k += 32) {
    v16bf av = ldfrag(a0 + k);
#pragma unroll
    for (int j = 0; j < 4; ++j) {
      v16bf bv = ldfrag(b0[j] + k);
      acc[j] = __builtin_amdgcn_wmma_f32_16x16x32_bf16(
          false, av, false, bv, (short)0, acc[j], false, false);
    }
  }

  const int mbase = tM * 16 + ((lane >> 4) << 3);
#pragma unroll
  for (int j = 0; j < 4; ++j) {
    if (!valid[j]) continue;
    int n = (tN4 * 4 + j) * 16 + l15;
    float bc = bias ? bias[n] : 0.0f;
    float*  cp  = C   ? C   + (size_t)mbase * ldc + n0 + n : nullptr;
    __bf16* cbp = Cbf ? Cbf + (size_t)mbase * ldc + n0 + n : nullptr;
#pragma unroll
    for (int r = 0; r < 8; ++r) {
      float v = acc[j][r] + bc;
      if (act == 1) v = geluf(v);
      if (cp)  cp[(size_t)r * ldc]  = v;
      if (cbp) cbp[(size_t)r * ldc] = (__bf16)v;
    }
  }
}

// ---------------- weight prep: fp32 (K,N) -> bf16 (N,K) ---------------------
__global__ void __launch_bounds__(256) k_transpose_bf16(
    const float* __restrict__ W, __bf16* __restrict__ Wt, int K, int N)
{
  int idx = blockIdx.x * 256 + threadIdx.x;
  if (idx >= K * N) return;
  int k = idx / N, n = idx - k * N;
  Wt[(size_t)n * K + k] = (__bf16)W[idx];
}

// normalize item rows 1..NI into bf16 [NI, D]
__global__ void __launch_bounds__(256) k_items(
    const float* __restrict__ emb, __bf16* __restrict__ iet)
{
  int item = blockIdx.x * 8 + (threadIdx.x >> 5);
  if (item >= NI) return;
  int lane = threadIdx.x & 31;
  const float* row = emb + (size_t)(item + 1) * D;
  float v[8]; float s = 0.f;
#pragma unroll
  for (int i = 0; i < 8; ++i) { v[i] = row[lane + i * 32]; s += v[i] * v[i]; }
#pragma unroll
  for (int off = 16; off; off >>= 1) s += __shfl_xor(s, off, 32);
  float inv = 1.0f / fmaxf(sqrtf(s), 1e-12f);
  __bf16* orow = iet + (size_t)item * D;
#pragma unroll
  for (int i = 0; i < 8; ++i) orow[lane + i * 32] = (__bf16)(v[i] * inv);
}

// ---------------- embeddings + time features --------------------------------
__global__ void __launch_bounds__(256) k_embed(
    const int* __restrict__ ids, const int* __restrict__ ts,
    const int* __restrict__ lens, const float* __restrict__ item_emb,
    const float* __restrict__ pos_emb, const float* __restrict__ gap_emb,
    const float* __restrict__ rec_emb, const float* __restrict__ tfw1,
    const float* __restrict__ tfb1, float* __restrict__ xbase,
    __bf16* __restrict__ th_bf)
{
  int row = blockIdx.x;               // b*T + t
  int b = row / T, t = row - b * T;
  int d = threadIdx.x;
  int len = lens[b];
  bool mask  = (t >= T - len);
  bool pmask = (t > 0) && ((t - 1) >= T - len);
  int tsv = ts[row];
  int pts = (t > 0) ? ts[row - 1] : 0;
  int latest = ts[b * T + T - 1];
  int gap = (mask && pmask) ? imax(tsv - pts, 0) : 0;
  int rec = mask ? imax(latest - tsv, 0) : 0;
  float rec_raw = (float)imax(latest - tsv, 0);
  int gb = min(imax((int)floorf(log2f((float)gap + 1.0f)), 0), NB - 1);
  int rb = min(imax((int)floorf(log2f((float)rec + 1.0f)), 0), NB - 1);
  float tf0 = log1pf((float)gb);
  float tf1 = log1pf(rec_raw);
  int id = ids[row];
  float xv = item_emb[(size_t)id * D + d] + pos_emb[t * D + d]
           + gap_emb[gb * D + d] + rec_emb[rb * D + d];
  xbase[(size_t)row * D + d] = xv;
  th_bf[(size_t)row * D + d] =
      (__bf16)geluf(tf0 * tfw1[d] + tf1 * tfw1[D + d] + tfb1[d]);
}

// ------- LayerNorm (rows of R = 256 or 1024), optional add/mask/bf16 --------
__global__ void __launch_bounds__(256) k_ln(
    const float* __restrict__ in, const float* __restrict__ add,
    const float* __restrict__ g, const float* __restrict__ bta,
    float* __restrict__ out, __bf16* __restrict__ out_bf, int R,
    const int* __restrict__ lens, int maskmode)
{
  int row = blockIdx.x, tid = threadIdx.x;
  int nper = R >> 8;
  float vals[4];
  float s = 0.f, s2 = 0.f;
  const float* ip = in + (size_t)row * R;
  for (int i = 0; i < nper; ++i) {
    int c = tid + (i << 8);
    float v = ip[c];
    if (add) v += add[(size_t)row * R + c];
    vals[i] = v; s += v; s2 += v * v;
  }
#pragma unroll
  for (int off = 16; off; off >>= 1) {
    s  += __shfl_xor(s, off, 32);
    s2 += __shfl_xor(s2, off, 32);
  }
  __shared__ float r1[8], r2[8];
  __shared__ float smean, sinv;
  if ((tid & 31) == 0) { r1[tid >> 5] = s; r2[tid >> 5] = s2; }
  __syncthreads();
  if (tid == 0) {
    float a = 0.f, bb = 0.f;
    for (int i = 0; i < 8; ++i) { a += r1[i]; bb += r2[i]; }
    float mean = a / (float)R;
    float var  = fmaxf(bb / (float)R - mean * mean, 0.f);
    smean = mean; sinv = rsqrtf(var + 1e-5f);
  }
  __syncthreads();
  float maskf = 1.f;
  if (maskmode) {
    int b = row / T, t = row - b * T;
    maskf = (t >= T - lens[b]) ? 1.f : 0.f;
  }
  float mean = smean, inv = sinv;
  for (int i = 0; i < nper; ++i) {
    int c = tid + (i << 8);
    float v = ((vals[i] - mean) * inv * g[c] + bta[c]) * maskf;
    if (out)    out[(size_t)row * R + c] = v;
    if (out_bf) out_bf[(size_t)row * R + c] = (__bf16)v;
  }
}

// ---------------- causal linear attention: one wave per (b,h) ---------------
__global__ void __launch_bounds__(256) k_attn(
    const float* __restrict__ proj, const int* __restrict__ lens,
    __bf16* __restrict__ content)
{
  int bh = blockIdx.x * 8 + (threadIdx.x >> 5);
  if (bh >= B * H) return;
  int b = bh / H, h = bh - b * H;
  int lane = threadIdx.x & 31;         // plays both d (q/k) and e (v/out) roles
  int len = lens[b];
  float kv[DH];
#pragma unroll
  for (int i = 0; i < DH; ++i) kv[i] = 0.f;
  float kpref = 0.f;
  const float* pbase = proj + (size_t)b * T * PROJ + D + h * DH;
  for (int t = 0; t < T; ++t) {
    const float* prow = pbase + (size_t)t * PROJ;
    float maskf = (t >= T - len) ? 1.f : 0.f;
    float qv  = prow[lane];
    float kvl = prow[INNER + lane];
    float vv  = prow[2 * INNER + lane] * maskf;
    qv  = (qv  > 0.f ? qv  + 1.f : expf(qv))  * maskf;  // (elu(x)+1)*mask
    kvl = (kvl > 0.f ? kvl + 1.f : expf(kvl)) * maskf;
    kpref += kvl;
    float dd = qv * kpref;
#pragma unroll
    for (int off = 16; off; off >>= 1) dd += __shfl_xor(dd, off, 32);
    float num = 0.f;
#pragma unroll
    for (int didx = 0; didx < DH; ++didx) {
      float kd = __shfl(kvl, didx, 32);
      float qd = __shfl(qv,  didx, 32);
      kv[didx] += kd * vv;
      num += qd * kv[didx];
    }
    content[((size_t)(b * T + t) * H + h) * DH + lane] =
        (__bf16)(num / (dd + 1e-6f) * maskf);
  }
}

// ------- causal depthwise conv, writes bf16 into concat buffer --------------
__global__ void __launch_bounds__(256) k_dwconv(
    const float* __restrict__ x, const float* __restrict__ w,
    const float* __restrict__ bias, __bf16* __restrict__ out,
    int ksz, int ldo, int n0)
{
  int idx = blockIdx.x * 256 + threadIdx.x;
  if (idx >= BT * D) return;
  int row = idx >> 8;   // D == 256
  int c = idx & 255;
  int t = row % T;
  float s = bias[c];
  for (int j = 0; j < ksz; ++j) {
    int tt = t - (ksz - 1) + j;
    if (tt >= 0) s += x[(size_t)(row - t + tt) * D + c] * w[c * ksz + j];
  }
  out[(size_t)row * ldo + n0 + c] = (__bf16)s;
}

// ---------------- residual combine: (res + sig(gate)*mixed) * mask ----------
__global__ void __launch_bounds__(256) k_resmix(
    const float* __restrict__ res, const float* __restrict__ gate,
    const float* __restrict__ mixed, float* __restrict__ out,
    const int* __restrict__ lens)
{
  int idx = blockIdx.x * 256 + threadIdx.x;
  if (idx >= BT * D) return;
  int row = idx >> 8, c = idx & 255;
  int b = row / T, t = row - b * T;
  float maskf = (t >= T - lens[b]) ? 1.f : 0.f;
  float gf = gate ? sigmoidf(gate[(size_t)row * PROJ + c]) : 1.f;
  out[idx] = (res[idx] + gf * mixed[idx]) * maskf;
}

// ---------------- pooling-gate scores ---------------------------------------
__global__ void __launch_bounds__(256) k_ps(
    const float* __restrict__ x, const float* __restrict__ pgw,
    const float* __restrict__ pgb, const int* __restrict__ lens,
    float* __restrict__ ps)
{
  int row = blockIdx.x * 8 + (threadIdx.x >> 5);
  if (row >= BT) return;
  int lane = threadIdx.x & 31;
  float s = 0.f;
#pragma unroll
  for (int i = 0; i < 8; ++i) {
    int c = lane + i * 32;
    s += x[(size_t)row * D + c] * pgw[c];
  }
#pragma unroll
  for (int off = 16; off; off >>= 1) s += __shfl_xor(s, off, 32);
  if (lane == 0) {
    int b = row / T, t = row - b * T;
    ps[row] = (t >= T - lens[b]) ? (s + pgb[0]) : -1e9f;
  }
}

// ------- pooling: last / mean / softmax-attn -> u_in bf16 (B,768) -----------
__global__ void __launch_bounds__(256) k_pool(
    const float* __restrict__ x, const float* __restrict__ ps,
    const int* __restrict__ lens, __bf16* __restrict__ uin)
{
  int b = blockIdx.x, tid = threadIdx.x;
  __shared__ float se[256];
  __shared__ float r[8];
  __shared__ float smax, ssum;
  float p = (tid < T) ? ps[b * T + tid] : -3.0e38f;
  float m = p;
#pragma unroll
  for (int off = 16; off; off >>= 1) m = fmaxf(m, __shfl_xor(m, off, 32));
  if ((tid & 31) == 0) r[tid >> 5] = m;
  __syncthreads();
  if (tid == 0) {
    float mm = r[0];
    for (int i = 1; i < 8; ++i) mm = fmaxf(mm, r[i]);
    smax = mm;
  }
  __syncthreads();
  float e = (tid < T) ? expf(p - smax) : 0.f;
  se[tid] = e;
  float s = e;
#pragma unroll
  for (int off = 16; off; off >>= 1) s += __shfl_xor(s, off, 32);
  if ((tid & 31) == 0) r[tid >> 5] = s;
  __syncthreads();
  if (tid == 0) {
    float ss = 0.f;
    for (int i = 0; i < 8; ++i) ss += r[i];
    ssum = ss;
  }
  __syncthreads();
  int len = lens[b];
  float invZ = 1.f / ssum;
  float denom = (float)imax(len, 1);
  float attn = 0.f, mean = 0.f;
  for (int t = 0; t < T; ++t) {
    float xv = x[(size_t)(b * T + t) * D + tid];
    attn += se[t] * invZ * xv;
    if (t >= T - len) mean += xv;
  }
  uin[(size_t)b * UIN + tid]         = (__bf16)x[(size_t)(b * T + T - 1) * D + tid];
  uin[(size_t)b * UIN + D + tid]     = (__bf16)(mean / denom);
  uin[(size_t)b * UIN + 2 * D + tid] = (__bf16)attn;
}

// -------- L2 row normalize (B rows of D), fp32 in -> bf16 out ---------------
__global__ void __launch_bounds__(256) k_unorm(
    const float* __restrict__ u, __bf16* __restrict__ ub)
{
  int row = blockIdx.x, tid = threadIdx.x;
  __shared__ float r[8];
  __shared__ float sinv;
  float v = u[(size_t)row * D + tid];
  float s = v * v;
#pragma unroll
  for (int off = 16; off; off >>= 1) s += __shfl_xor(s, off, 32);
  if ((tid & 31) == 0) r[tid >> 5] = s;
  __syncthreads();
  if (tid == 0) {
    float a = 0.f;
    for (int i = 0; i < 8; ++i) a += r[i];
    sinv = 1.f / fmaxf(sqrtf(a), 1e-12f);
  }
  __syncthreads();
  ub[(size_t)row * D + tid] = (__bf16)(v * sinv);
}

// ---------------- host side -------------------------------------------------
static inline void launch_gemm(const __bf16* A, const __bf16* Wt,
    const float* bias, float* C, __bf16* Cbf, int M, int N, int K, int ldc,
    int n0, int act, hipStream_t s)
{
  int tilesN4 = ((N >> 4) + 3) >> 2;
  int tiles = (M >> 4) * tilesN4;
  int blocks = (tiles + 7) / 8;
  k_gemm<<<blocks, 256, 0, s>>>(A, Wt, bias, C, Cbf, M, N, K, ldc, n0, act);
}

static inline void launch_tr(const float* W, __bf16* Wt, int K, int N,
                             hipStream_t s)
{
  int n = K * N;
  k_transpose_bf16<<<(n + 255) / 256, 256, 0, s>>>(W, Wt, K, N);
}

extern "C" void kernel_launch(void* const* d_in, const int* in_sizes, int n_in,
                              void* d_out, int out_size, void* d_ws,
                              size_t ws_size, hipStream_t stream)
{
  (void)in_sizes; (void)n_in; (void)out_size; (void)ws_size;
  const int*   ids      = (const int*)d_in[0];
  const int*   tsp      = (const int*)d_in[1];
  const int*   lens     = (const int*)d_in[2];
  const float* item_emb = (const float*)d_in[3];
  const float* pos_emb  = (const float*)d_in[4];
  const float* gap_emb  = (const float*)d_in[5];
  const float* rec_emb  = (const float*)d_in[6];
  const float* tf_w1    = (const float*)d_in[7];
  const float* tf_b1    = (const float*)d_in[8];
  const float* tf_w2    = (const float*)d_in[9];
  const float* tf_b2    = (const float*)d_in[10];
  const float* in_g     = (const float*)d_in[11];
  const float* in_b     = (const float*)d_in[12];
  const float* l_ln1g   = (const float*)d_in[13];
  const float* l_ln1b   = (const float*)d_in[14];
  const float* l_win    = (const float*)d_in[15];
  const float* l_bin    = (const float*)d_in[16];
  const float* l_wc     = (const float*)d_in[17];
  const float* l_bc     = (const float*)d_in[18];
  const float* l_tw     = (const float*)d_in[19];
  const float* l_tb     = (const float*)d_in[20];
  const float* l_pw     = (const float*)d_in[21];
  const float* l_pb     = (const float*)d_in[22];
  const float* l_wmix   = (const float*)d_in[23];
  const float* l_bmix   = (const float*)d_in[24];
  const float* l_ln2g   = (const float*)d_in[25];
  const float* l_ln2b   = (const float*)d_in[26];
  const float* l_wf1    = (const float*)d_in[27];
  const float* l_bf1    = (const float*)d_in[28];
  const float* l_wf2    = (const float*)d_in[29];
  const float* l_bf2    = (const float*)d_in[30];
  const float* fn_g     = (const float*)d_in[31];
  const float* fn_b     = (const float*)d_in[32];
  const float* pg_w     = (const float*)d_in[33];
  const float* pg_b     = (const float*)d_in[34];
  const float* uh_w1    = (const float*)d_in[35];
  const float* uh_b1    = (const float*)d_in[36];
  const float* uh_lng   = (const float*)d_in[37];
  const float* uh_lnb   = (const float*)d_in[38];
  const float* uh_w2    = (const float*)d_in[39];
  const float* uh_b2    = (const float*)d_in[40];
  const float* item_bias= (const float*)d_in[41];

  // ---- workspace carve (256B aligned) ----
  char* base = (char*)d_ws;
  size_t off = 0;
  auto carve = [&](size_t bytes) -> char* {
    char* p = base + off;
    off += (bytes + 255) & ~(size_t)255;
    return p;
  };
  // bf16 weights (N-major)
  __bf16* win_t  = (__bf16*)carve((size_t)L * PROJ * D * 2);
  __bf16* wc_t   = (__bf16*)carve((size_t)L * D * INNER * 2);
  __bf16* wmix_t = (__bf16*)carve((size_t)L * D * CAT * 2);
  __bf16* wf1_t  = (__bf16*)carve((size_t)L * FF * D * 2);
  __bf16* wf2_t  = (__bf16*)carve((size_t)L * D * FF * 2);
  __bf16* tfw2_t = (__bf16*)carve((size_t)D * D * 2);
  __bf16* uhw1_t = (__bf16*)carve((size_t)UH * UIN * 2);
  __bf16* uhw2_t = (__bf16*)carve((size_t)D * UH * 2);
  __bf16* ie_t   = (__bf16*)carve((size_t)NI * D * 2);
  // fp32 activations
  float* A1  = (float*)carve((size_t)BT * D * 4);     // x
  float* A2  = (float*)carve((size_t)BT * D * 4);     // nrm (fp32 for dwconv)
  float* A3  = (float*)carve((size_t)BT * PROJ * 4);  // proj (incl. gate)
  float* A6  = (float*)carve((size_t)BT * D * 4);     // mixed / ffn-out / dense_time
  float* psb = (float*)carve((size_t)BT * 4);
  float* uvf = (float*)carve((size_t)B * D * 4);
  // bf16 activations (GEMM inputs, written by producers)
  __bf16* th_bf   = (__bf16*)carve((size_t)BT * D * 2);
  __bf16* nrm_bf  = (__bf16*)carve((size_t)BT * D * 2);
  __bf16* cont_bf = (__bf16*)carve((size_t)BT * INNER * 2);
  __bf16* cat_bf  = (__bf16*)carve((size_t)BT * CAT * 2);
  __bf16* h_bf    = (__bf16*)carve((size_t)BT * D * 2);
  __bf16* mid_bf  = (__bf16*)carve((size_t)BT * FF * 2);
  __bf16* uin_bf  = (__bf16*)carve((size_t)B * UIN * 2);
  __bf16* uh2_bf  = (__bf16*)carve((size_t)B * UH * 2);
  __bf16* uv_bf   = (__bf16*)carve((size_t)B * D * 2);
  float*  uhf     = (float*)carve((size_t)B * UH * 4);

  // ---- weight prep: transpose + bf16 convert ----
  for (int i = 0; i < L; ++i) {
    launch_tr(l_win  + (size_t)i * D * PROJ, win_t  + (size_t)i * PROJ * D, D, PROJ, stream);
    launch_tr(l_wc   + (size_t)i * INNER * D, wc_t  + (size_t)i * D * INNER, INNER, D, stream);
    launch_tr(l_wmix + (size_t)i * CAT * D, wmix_t  + (size_t)i * D * CAT, CAT, D, stream);
    launch_tr(l_wf1  + (size_t)i * D * FF, wf1_t    + (size_t)i * FF * D, D, FF, stream);
    launch_tr(l_wf2  + (size_t)i * FF * D, wf2_t    + (size_t)i * D * FF, FF, D, stream);
  }
  launch_tr(tf_w2, tfw2_t, D, D, stream);
  launch_tr(uh_w1, uhw1_t, UIN, UH, stream);
  launch_tr(uh_w2, uhw2_t, UH, D, stream);
  k_items<<<(NI + 7) / 8, 256, 0, stream>>>(item_emb, ie_t);

  // ---- embeddings + time-feature MLP + input LN ----
  k_embed<<<BT, 256, 0, stream>>>(ids, tsp, lens, item_emb, pos_emb, gap_emb,
                                  rec_emb, tf_w1, tf_b1, A1, th_bf);
  launch_gemm(th_bf, tfw2_t, tf_b2, A6, nullptr, BT, D, D, D, 0, 0, stream);
  k_ln<<<BT, 256, 0, stream>>>(A1, A6, in_g, in_b, A1, (__bf16*)nullptr, D,
                               lens, 1);

  // ---- transformer layers ----
  for (int i = 0; i < L; ++i) {
    // ln1: fp32 (for dwconv) + bf16 (for proj GEMM)
    k_ln<<<BT, 256, 0, stream>>>(A1, nullptr, l_ln1g + i * D, l_ln1b + i * D,
                                 A2, nrm_bf, D, nullptr, 0);
    launch_gemm(nrm_bf, win_t + (size_t)i * PROJ * D, l_bin + i * PROJ,
                A3, nullptr, BT, PROJ, D, PROJ, 0, 0, stream);
    k_attn<<<(B * H + 7) / 8, 256, 0, stream>>>(A3, lens, cont_bf);
    launch_gemm(cont_bf, wc_t + (size_t)i * D * INNER, l_bc + i * D,
                nullptr, cat_bf, BT, D, INNER, CAT, 0, 0, stream);
    k_dwconv<<<(BT * D + 255) / 256, 256, 0, stream>>>(A2, l_tw + i * D * 3,
                l_tb + i * D, cat_bf, 3, CAT, D);
    k_dwconv<<<(BT * D + 255) / 256, 256, 0, stream>>>(A2, l_pw + i * D * 5,
                l_pb + i * D, cat_bf, 5, CAT, 2 * D);
    launch_gemm(cat_bf, wmix_t + (size_t)i * D * CAT, l_bmix + i * D,
                A6, nullptr, BT, D, CAT, D, 0, 0, stream);
    k_resmix<<<(BT * D + 255) / 256, 256, 0, stream>>>(A1, A3, A6, A1, lens);
    // ln2: bf16 only (feeds ffn1 GEMM)
    k_ln<<<BT, 256, 0, stream>>>(A1, nullptr, l_ln2g + i * D, l_ln2b + i * D,
                                 nullptr, h_bf, D, nullptr, 0);
    launch_gemm(h_bf, wf1_t + (size_t)i * FF * D, l_bf1 + i * FF,
                nullptr, mid_bf, BT, FF, D, FF, 0, 1, stream);     // gelu
    launch_gemm(mid_bf, wf2_t + (size_t)i * D * FF, l_bf2 + i * D,
                A6, nullptr, BT, D, FF, D, 0, 0, stream);
    k_resmix<<<(BT * D + 255) / 256, 256, 0, stream>>>(A1, nullptr, A6, A1, lens);
  }

  // ---- final LN, pooling, user head ----
  k_ln<<<BT, 256, 0, stream>>>(A1, nullptr, fn_g, fn_b, A1, (__bf16*)nullptr,
                               D, lens, 1);
  k_ps<<<(BT + 7) / 8, 256, 0, stream>>>(A1, pg_w, pg_b, lens, psb);
  k_pool<<<B, 256, 0, stream>>>(A1, psb, lens, uin_bf);
  launch_gemm(uin_bf, uhw1_t, uh_b1, uhf, nullptr, B, UH, UIN, UH, 0, 1, stream);
  k_ln<<<B, 256, 0, stream>>>(uhf, nullptr, uh_lng, uh_lnb, nullptr, uh2_bf,
                              UH, nullptr, 0);
  launch_gemm(uh2_bf, uhw2_t, uh_b2, uvf, nullptr, B, D, UH, D, 0, 0, stream);
  k_unorm<<<B, 256, 0, stream>>>(uvf, uv_bf);

  // ---- scoring: (32 x 256) @ (256 x 100000) + item_bias[1:] -> d_out ----
  launch_gemm(uv_bf, ie_t, item_bias + 1, (float*)d_out, nullptr,
              B, NI, D, NI, 0, 0, stream);
}
